// MLP_7791070675691
// MI455X (gfx1250) — matrix operand order repaired and checked by reading, not compiled, not run on hardware
//
#include <hip/hip_runtime.h>

// MI455X / gfx1250 MLP: 1M x (64->32->12->8->6->2), fp32 in/out.
// Roofline: 264 MB of mandatory HBM traffic @ 23.3 TB/s ~ 11.3 us; AI ~ 19.6
// F/B means fp16 WMMA (f32 accumulate) keeps compute far below the memory
// floor. This version removes the round-1 bottleneck (per-wave scalar weight
// loads ~ 1 GB of L2 traffic): weights/biases are padded+converted once per
// block into LDS in the exact WMMA B layout, register-cached per wave, and
// each wave streams 4 x 16-row tiles with prefetched b128 x loads.

typedef __attribute__((ext_vector_type(16))) _Float16 v16h;
typedef __attribute__((ext_vector_type(8)))  _Float16 v8h;
typedef __attribute__((ext_vector_type(8)))  float    v8f;
typedef __attribute__((ext_vector_type(4)))  float    f32x4;

#define WAVES 8   // waves per block
#define TPW   4   // 16-row tiles per wave -> 512 rows per block

// Same-wave LDS store->load ordering (DS pipe is in-order per wave; the wait
// plus the memory clobber keeps the compiler honest too).
#define LDS_FENCE() asm volatile("s_wait_dscnt 0" ::: "memory")

static __device__ __forceinline__ v8f wmma_f16(v16h a, v16h b, v8f c) {
    // 8 args: (neg_a, A, neg_b, B, c_mod, C, reuse_a, reuse_b)
    return __builtin_amdgcn_wmma_f32_16x16x32_f16(false, a, false, b,
                                                  (short)0, c, false, false);
}

static __device__ __forceinline__ v8f splat8(float s) {
    v8f c;
#pragma unroll
    for (int v = 0; v < 8; ++v) c[v] = s;
    return c;
}

static __device__ __forceinline__ v8f relu8(v8f c) {
#pragma unroll
    for (int v = 0; v < 8; ++v) c[v] = fmaxf(c[v], 0.0f);
    return c;
}

// Cooperative prep: one 32x16 (KxN) B tile, padded with zeros, stored to LDS
// in the per-lane WMMA B layout (dst[l*16+i]; lane l holds column n = l&15,
// element i -> k = 8*(l>>4) + i + 8*(i>>3)). 256 threads fill 512 halves (2 each).
static __device__ __forceinline__ void prep_tileB(_Float16* dst,
                                                  const float* __restrict__ W,
                                                  int Kw, int Nw, int kOff,
                                                  int nOff, int tid) {
    const int l  = tid >> 3;          // 0..31 (B-layout lane)
    const int i0 = (tid & 7) << 1;    // element pair
    const int n  = (l & 15) + nOff;
    const int kb = kOff + ((l >> 4) << 3);
#pragma unroll
    for (int j = 0; j < 2; ++j) {
        const int i = i0 + j;
        const int k = kb + i + ((i >> 3) << 3);
        const float w = (k < Kw && n < Nw) ? W[k * Nw + n] : 0.0f;
        dst[l * 16 + i] = (_Float16)w;
    }
}

// Store C-layout accumulator (lane: N = lane&15 (+nOff), M = v + 8*(lane>>4))
// into a row-major 16x32 f16 tile in LDS.
static __device__ __forceinline__ void store_tile(_Float16* h, v8f c,
                                                  int nOff, int lane) {
    const int n  = (lane & 15) + nOff;
    const int m0 = (lane >> 4) << 3;
#pragma unroll
    for (int v = 0; v < 8; ++v) h[(m0 + v) * 32 + n] = (_Float16)c[v];
}

// Read the row-major 16x32 f16 tile back in the 16-bit A layout.
static __device__ __forceinline__ v16h load_A(const _Float16* h, int lane,
                                              bool full) {
    const int row = lane & 15;
    const int k0  = (lane >> 4) << 3;
    v16h a;
    const v8h lo = *(const v8h*)(h + row * 32 + k0);          // ds_load_b128
#pragma unroll
    for (int i = 0; i < 8; ++i) a[i] = lo[i];
    if (full) {
        const v8h hp = *(const v8h*)(h + row * 32 + 16 + k0);
#pragma unroll
        for (int i = 0; i < 8; ++i) a[8 + i] = hp[i];
    } else {
#pragma unroll
        for (int i = 0; i < 8; ++i) a[8 + i] = (_Float16)0.0f;
    }
    return a;
}

// 8 x global_load_b128: one lane's share of a 16x64 x tile (32 floats).
static __device__ __forceinline__ void load_x8(const float* __restrict__ p,
                                               f32x4* r) {
    r[0] = *(const f32x4*)(p + 0);   r[1] = *(const f32x4*)(p + 4);
    r[2] = *(const f32x4*)(p + 16);  r[3] = *(const f32x4*)(p + 20);
    r[4] = *(const f32x4*)(p + 32);  r[5] = *(const f32x4*)(p + 36);
    r[6] = *(const f32x4*)(p + 48);  r[7] = *(const f32x4*)(p + 52);
}

// Convert one lane's 32 fp32 x values into two A-layout v16h (K 0..31, 32..63).
static __device__ __forceinline__ void cvt_a(const f32x4* r, v16h& a0, v16h& a1) {
#pragma unroll
    for (int j = 0; j < 4; ++j) {
        a0[j]      = (_Float16)r[0][j];  a0[4 + j]  = (_Float16)r[1][j];
        a0[8 + j]  = (_Float16)r[2][j];  a0[12 + j] = (_Float16)r[3][j];
        a1[j]      = (_Float16)r[4][j];  a1[4 + j]  = (_Float16)r[5][j];
        a1[8 + j]  = (_Float16)r[6][j];  a1[12 + j] = (_Float16)r[7][j];
    }
}

__global__ __launch_bounds__(256) void MLP_7791070675691_kernel(
    const float* __restrict__ x,
    const float* __restrict__ W0, const float* __restrict__ b0,
    const float* __restrict__ W1, const float* __restrict__ b1,
    const float* __restrict__ W2, const float* __restrict__ b2,
    const float* __restrict__ W3, const float* __restrict__ b3,
    const float* __restrict__ W4, const float* __restrict__ b4,
    float* __restrict__ out)
{
    __shared__ __align__(32) _Float16 sB[8 * 512];        // 8 KB: B tiles
    __shared__ float sBias[6 * 32];                       // padded bias rows
    __shared__ __align__(32) _Float16 sH[WAVES][512];     // per-wave h bounce

    const int tid = threadIdx.x;

    // ---- Once per block: build padded f16 weight tiles in WMMA B layout ----
    prep_tileB(sB + 0 * 512, W0, 64, 32,  0,  0, tid);   // L0  K 0..31, N 0..15
    prep_tileB(sB + 1 * 512, W0, 64, 32, 32,  0, tid);   // L0  K32..63, N 0..15
    prep_tileB(sB + 2 * 512, W0, 64, 32,  0, 16, tid);   // L0  K 0..31, N16..31
    prep_tileB(sB + 3 * 512, W0, 64, 32, 32, 16, tid);   // L0  K32..63, N16..31
    prep_tileB(sB + 4 * 512, W1, 32, 12,  0,  0, tid);   // L1
    prep_tileB(sB + 5 * 512, W2, 12,  8,  0,  0, tid);   // L2
    prep_tileB(sB + 6 * 512, W3,  8,  6,  0,  0, tid);   // L3
    prep_tileB(sB + 7 * 512, W4,  6,  2,  0,  0, tid);   // L4
    if (tid < 192) {                                     // 6 bias rows x 32
        const int t = tid >> 5, l = tid & 31;
        const float* bp = b0; int Nw = 0, nOff = 0;
        switch (t) {
            case 0: bp = b0; Nw = 32; nOff = 0;  break;
            case 1: bp = b0; Nw = 32; nOff = 16; break;
            case 2: bp = b1; Nw = 12; break;
            case 3: bp = b2; Nw = 8;  break;
            case 4: bp = b3; Nw = 6;  break;
            case 5: bp = b4; Nw = 2;  break;
        }
        const int n = (l & 15) + nOff;
        sBias[t * 32 + l] = (n < Nw) ? bp[n] : 0.0f;
    }
    __syncthreads();

    const int lane = tid & 31;
    const int wave = tid >> 5;
    const int row  = lane & 15;
    const int hi   = lane >> 4;
    _Float16* h = &sH[wave][0];

    // ---- Once per wave: register-cache B tiles (2 x ds_load_b128 each) ----
    const v16h Bk0n0 = *(const v16h*)(sB + 0 * 512 + lane * 16);
    const v16h Bk1n0 = *(const v16h*)(sB + 1 * 512 + lane * 16);
    const v16h Bk0n1 = *(const v16h*)(sB + 2 * 512 + lane * 16);
    const v16h Bk1n1 = *(const v16h*)(sB + 3 * 512 + lane * 16);
    const v16h B1    = *(const v16h*)(sB + 4 * 512 + lane * 16);
    const v16h B2    = *(const v16h*)(sB + 5 * 512 + lane * 16);
    const v16h B3    = *(const v16h*)(sB + 6 * 512 + lane * 16);
    const v16h B4    = *(const v16h*)(sB + 7 * 512 + lane * 16);
    const float bb00 = sBias[0 * 32 + lane];
    const float bb01 = sBias[1 * 32 + lane];
    const float bb1  = sBias[2 * 32 + lane];
    const float bb2  = sBias[3 * 32 + lane];
    const float bb3  = sBias[4 * 32 + lane];
    const float bb4  = sBias[5 * 32 + lane];

    // ---- Stream TPW 16-row tiles, prefetching the next tile's x ----
    const int tileBase = (blockIdx.x * WAVES + wave) * TPW;
    const float* xw = x + ((size_t)tileBase * 16 + row) * 64 + hi * 8;

    f32x4 pre[8];
    load_x8(xw, pre);

#pragma unroll
    for (int it = 0; it < TPW; ++it) {
        v16h a0, a1;
        cvt_a(pre, a0, a1);
        if (it + 1 < TPW) load_x8(xw + (size_t)(it + 1) * 16 * 64, pre);

        // Layer 0: 64 -> 32 (2 K-chunks x 2 N-tiles)
        v8f c0 = splat8(bb00);
        v8f c1 = splat8(bb01);
        c0 = wmma_f16(a0, Bk0n0, c0);
        c0 = wmma_f16(a1, Bk1n0, c0);
        c1 = wmma_f16(a0, Bk0n1, c1);
        c1 = wmma_f16(a1, Bk1n1, c1);
        c0 = relu8(c0);
        c1 = relu8(c1);
        store_tile(h, c0, 0, lane);
        store_tile(h, c1, 16, lane);
        LDS_FENCE();

        // Layer 1: 32 -> 12 (padded output cols are exact zeros)
        v16h a = load_A(h, lane, true);
        v8f  c = splat8(bb1);
        c = wmma_f16(a, B1, c);
        c = relu8(c);
        store_tile(h, c, 0, lane);
        LDS_FENCE();

        // Layer 2: 12 -> 8 (K <= 16: high half of A zeroed in registers)
        a = load_A(h, lane, false);
        c = splat8(bb2);
        c = wmma_f16(a, B2, c);
        c = relu8(c);
        store_tile(h, c, 0, lane);
        LDS_FENCE();

        // Layer 3: 8 -> 6
        a = load_A(h, lane, false);
        c = splat8(bb3);
        c = wmma_f16(a, B3, c);
        c = relu8(c);
        store_tile(h, c, 0, lane);
        LDS_FENCE();

        // Layer 4: 6 -> 2 (no ReLU)
        a = load_A(h, lane, false);
        c = splat8(bb4);
        c = wmma_f16(a, B4, c);

        // Coalesced fp32 output via the LDS tile (reused as float scratch)
        float* ob = (float*)h;
        if ((lane & 15) < 2) {
            const int n  = lane & 15;
            const int m0 = hi << 3;
#pragma unroll
            for (int v = 0; v < 8; ++v) ob[(m0 + v) * 2 + n] = c[v];
        }
        LDS_FENCE();
        out[((size_t)(tileBase + it)) * 32 + lane] = ob[lane];
        LDS_FENCE();   // ob reads land before next iteration's h stores
    }
}

extern "C" void kernel_launch(void* const* d_in, const int* in_sizes, int n_in,
                              void* d_out, int out_size, void* d_ws, size_t ws_size,
                              hipStream_t stream) {
    const float* x  = (const float*)d_in[0];
    const float* W0 = (const float*)d_in[1];  const float* b0 = (const float*)d_in[2];
    const float* W1 = (const float*)d_in[3];  const float* b1 = (const float*)d_in[4];
    const float* W2 = (const float*)d_in[5];  const float* b2 = (const float*)d_in[6];
    const float* W3 = (const float*)d_in[7];  const float* b3 = (const float*)d_in[8];
    const float* W4 = (const float*)d_in[9];  const float* b4 = (const float*)d_in[10];
    float* out = (float*)d_out;

    const int nrows  = in_sizes[0] / 64;             // 1,048,576
    const int blocks = nrows / (WAVES * TPW * 16);   // 512 rows/block -> 2048

    MLP_7791070675691_kernel<<<blocks, 256, 0, stream>>>(
        x, W0, b0, W1, b1, W2, b2, W3, b3, W4, b4, out);
}
